// Matformer_AUG_80917183857167
// MI455X (gfx1250) — matrix operand order repaired and checked by reading, not compiled, not run on hardware
//
#include <hip/hip_runtime.h>
#include <math.h>

typedef __attribute__((ext_vector_type(16))) _Float16 v16h;
typedef __attribute__((ext_vector_type(8)))  float    v8f;

struct alignas(16) U4 { unsigned int a, b, c, d; };
union Frag { v16h v; U4 q[2]; };

__device__ __forceinline__ float sigmoid_f(float x) { return 1.0f / (1.0f + expf(-x)); }

// ---------------------------------------------------------------------------
// Small helper kernels
// ---------------------------------------------------------------------------
__global__ void zero_kernel(float* p, int n) {
    int i = blockIdx.x * 256 + threadIdx.x;
    if (i < n) p[i] = 0.0f;
}

__global__ void f32_to_f16_kernel(const float* __restrict__ s, _Float16* __restrict__ d, int n) {
    int i = blockIdx.x * 256 + threadIdx.x;
    if (i < n) d[i] = (_Float16)s[i];
}

// src [K,N] f32 -> dst [N,Kp] f16 (transposed, K padded with zeros)
__global__ void convert_T_pad_kernel(const float* __restrict__ s, _Float16* __restrict__ d,
                                     int K, int Nn, int Kp) {
    int i = blockIdx.x * 256 + threadIdx.x;
    if (i >= Nn * Kp) return;
    int n = i / Kp, kk = i - n * Kp;
    d[i] = (kk < K) ? (_Float16)s[(size_t)kk * Nn + n] : (_Float16)0.0f;
}

// src [M,K] f32 -> dst [M,Kp] f16 (row-major, K padded with zeros)
__global__ void convert_pad_rows_kernel(const float* __restrict__ s, _Float16* __restrict__ d,
                                        int M, int K, int Kp) {
    int i = blockIdx.x * 256 + threadIdx.x;
    if (i >= M * Kp) return;
    int r = i / Kp, kk = i - r * Kp;
    d[i] = (kk < K) ? (_Float16)s[(size_t)r * K + kk] : (_Float16)0.0f;
}

// ---------------------------------------------------------------------------
// Generic WMMA GEMM:  out[M,N] = A[M,K](f16) * BT[N,K](f16)^T + bias
// grid = (M/16, N/64), block = 128 (4 waves, wave w -> column tile w)
// ---------------------------------------------------------------------------
template <typename OutT>
__global__ __launch_bounds__(128) void gemm_wmma_kernel(
    const _Float16* __restrict__ A, const _Float16* __restrict__ BT,
    const float* __restrict__ bias, OutT* __restrict__ out,
    int M, int N, int K) {
    const int lane = threadIdx.x & 31;
    const int wave = threadIdx.x >> 5;
    const int row0 = blockIdx.x * 16;
    const int col0 = blockIdx.y * 64 + wave * 16;

    const int arow = row0 + (lane & 15);
    const _Float16* ap = A + (size_t)arow * K + ((lane < 16) ? 0 : 8);
    const int bcol = col0 + (lane & 15);
    const _Float16* bp = BT + (size_t)bcol * K + ((lane < 16) ? 0 : 8);

    v8f acc{};
    for (int kb = 0; kb < K; kb += 32) {
        Frag af, bf;
        af.q[0] = *(const U4*)(ap + kb);
        af.q[1] = *(const U4*)(ap + kb + 16);
        bf.q[0] = *(const U4*)(bp + kb);
        bf.q[1] = *(const U4*)(bp + kb + 16);
        if (kb + 32 < K) __builtin_prefetch((const void*)(bp + kb + 32), 0, 1);
        acc = __builtin_amdgcn_wmma_f32_16x16x32_f16(false, af.v, false, bf.v,
                                                     (short)0, acc, false, false);
    }
    const int n = col0 + (lane & 15);
    const float bs = bias ? bias[n] : 0.0f;
    const int mbase = row0 + ((lane < 16) ? 0 : 8);
#pragma unroll
    for (int i = 0; i < 8; ++i)
        out[(size_t)(mbase + i) * N + n] = (OutT)(acc[i] + bs);
}

// ---------------------------------------------------------------------------
// Fused edge-message kernel (per layer).
// Rows g = e*H + h  (H=2). Block handles 32 rows (16 edges x 2 heads).
// Phase A: gather A1 = f16[concat(v[dst],v[src],ea)] and alpha = q*k*scale (LDS)
// Phase B: gate = sigmoid(LN_384(alpha))              (LDS, in-place)
// Phase C: S = f16((A1 @ WmuT^T + bmu) * gate)        (WMMA, LDS)
// Phase D: D2 = S @ WmT^T + bm                        (WMMA, LDS)
// Phase E: LN_128(D2) -> atomicAdd into agg[dst]
// Dynamic LDS: 24K(A1) + 48K(G) + 24K(S) + 16K(D2) = 112 KB
// ---------------------------------------------------------------------------
__global__ __launch_bounds__(256) void edge_msg_kernel(
    const float* __restrict__ q, const float* __restrict__ k, const float* __restrict__ v,
    const _Float16* __restrict__ ea, const int* __restrict__ eidx,
    const _Float16* __restrict__ WmuT, const float* __restrict__ bmu,
    const _Float16* __restrict__ WmT, const float* __restrict__ bm,
    const float* __restrict__ lnag, const float* __restrict__ lnab,
    const float* __restrict__ lnmg, const float* __restrict__ lnmb,
    float* __restrict__ agg, int E) {
    extern __shared__ char lds[];
    _Float16* A1 = (_Float16*)lds;                          // [32][384] f16
    float*    G  = (float*)(lds + 24576);                   // [32][384] f32
    _Float16* S  = (_Float16*)(lds + 24576 + 49152);        // [32][384] f16
    float*    D2 = (float*)(lds + 24576 + 49152 + 24576);   // [32][128] f32

    const int tid  = threadIdx.x;
    const int lane = tid & 31;
    const int wave = tid >> 5;
    const int g0 = blockIdx.x * 32;
    const float scale = 0.05103103630798287f;  // 1/sqrt(384)

    // ---- Phase A: gather ----
    for (int idx = tid; idx < 32 * 384; idx += 256) {
        int r = idx / 384, f = idx - r * 384;
        int g = g0 + r, e = g >> 1, h = g & 1;
        int sidx = eidx[e], didx = eidx[E + e];
        int c = f & 127, seg = f >> 7;
        int hoff = h * 128 + c;
        float av, kv;
        if (seg == 0)      { av = v[(size_t)didx * 256 + hoff]; kv = k[(size_t)didx * 256 + hoff]; }
        else if (seg == 1) { av = v[(size_t)sidx * 256 + hoff]; kv = k[(size_t)sidx * 256 + hoff]; }
        else               { av = (float)ea[(size_t)e * 256 + hoff]; kv = av; }
        A1[r * 384 + f] = (_Float16)av;
        float qv = q[(size_t)didx * 256 + hoff];
        G[r * 384 + f] = qv * kv * scale;
    }
    __syncthreads();

    // ---- Phase B: gate = sigmoid(LN_384(alpha)) ----
    for (int r = wave * 4; r < wave * 4 + 4; ++r) {
        float s = 0.0f, s2 = 0.0f;
        for (int j = 0; j < 12; ++j) {
            float a = G[r * 384 + lane + j * 32];
            s += a; s2 += a * a;
        }
        for (int o = 16; o > 0; o >>= 1) { s += __shfl_xor(s, o); s2 += __shfl_xor(s2, o); }
        float mn = s * (1.0f / 384.0f);
        float var = s2 * (1.0f / 384.0f) - mn * mn;
        float rs = rsqrtf(var + 1e-5f);
        for (int j = 0; j < 12; ++j) {
            int f = lane + j * 32;
            float a = G[r * 384 + f];
            G[r * 384 + f] = sigmoid_f((a - mn) * rs * lnag[f] + lnab[f]);
        }
    }
    __syncthreads();

    // ---- Phase C: S = f16((A1 @ WmuT^T + bmu) * gate). 48 tiles / 8 waves ----
    {
        const int acol = lane & 15;
        const int ak = (lane < 16) ? 0 : 8;
        const int bk = (lane < 16) ? 0 : 8;
        for (int tt = wave; tt < 48; tt += 8) {
            int rt = tt & 1, ct = tt >> 1;
            int n0 = ct * 16;
            const _Float16* ap = A1 + (rt * 16 + acol) * 384 + ak;
            const _Float16* bp = WmuT + (size_t)(n0 + acol) * 384 + bk;
            v8f acc{};
            for (int kb = 0; kb < 384; kb += 32) {
                Frag af, bf;
                af.q[0] = *(const U4*)(ap + kb);
                af.q[1] = *(const U4*)(ap + kb + 16);
                bf.q[0] = *(const U4*)(bp + kb);
                bf.q[1] = *(const U4*)(bp + kb + 16);
                acc = __builtin_amdgcn_wmma_f32_16x16x32_f16(false, af.v, false, bf.v,
                                                             (short)0, acc, false, false);
            }
            int mrow0 = rt * 16 + ((lane < 16) ? 0 : 8);
            int ncol = n0 + acol;
            float bb = bmu[ncol];
#pragma unroll
            for (int i = 0; i < 8; ++i) {
                int rr = mrow0 + i;
                S[rr * 384 + ncol] = (_Float16)((acc[i] + bb) * G[rr * 384 + ncol]);
            }
        }
    }
    __syncthreads();

    // ---- Phase D: D2 = S @ WmT^T + bm. 16 tiles / 8 waves ----
    {
        const int acol = lane & 15;
        const int ak = (lane < 16) ? 0 : 8;
        const int bk = (lane < 16) ? 0 : 8;
        for (int tt = wave; tt < 16; tt += 8) {
            int rt = tt & 1, ct = tt >> 1;
            int n0 = ct * 16;
            const _Float16* ap = S + (rt * 16 + acol) * 384 + ak;
            const _Float16* bp = WmT + (size_t)(n0 + acol) * 384 + bk;
            v8f acc{};
            for (int kb = 0; kb < 384; kb += 32) {
                Frag af, bf;
                af.q[0] = *(const U4*)(ap + kb);
                af.q[1] = *(const U4*)(ap + kb + 16);
                bf.q[0] = *(const U4*)(bp + kb);
                bf.q[1] = *(const U4*)(bp + kb + 16);
                acc = __builtin_amdgcn_wmma_f32_16x16x32_f16(false, af.v, false, bf.v,
                                                             (short)0, acc, false, false);
            }
            int mrow0 = rt * 16 + ((lane < 16) ? 0 : 8);
            int ncol = n0 + acol;
            float bb = bm[ncol];
#pragma unroll
            for (int i = 0; i < 8; ++i)
                D2[(mrow0 + i) * 128 + ncol] = acc[i] + bb;
        }
    }
    __syncthreads();

    // ---- Phase E: LN_128 + scatter-add ----
    for (int r = wave * 4; r < wave * 4 + 4; ++r) {
        float s = 0.0f, s2 = 0.0f;
        for (int j = 0; j < 4; ++j) {
            float a = D2[r * 128 + lane + j * 32];
            s += a; s2 += a * a;
        }
        for (int o = 16; o > 0; o >>= 1) { s += __shfl_xor(s, o); s2 += __shfl_xor(s2, o); }
        float mn = s * (1.0f / 128.0f);
        float var = s2 * (1.0f / 128.0f) - mn * mn;
        float rs = rsqrtf(var + 1e-5f);
        int g = g0 + r, e = g >> 1, h = g & 1;
        int didx = eidx[E + e];
        for (int j = 0; j < 4; ++j) {
            int c = lane + j * 32;
            float val = (D2[r * 128 + c] - mn) * rs * lnmg[c] + lnmb[c];
            atomicAdd(&agg[(size_t)didx * 256 + h * 128 + c], val);
        }
    }
}

// ---------------------------------------------------------------------------
// BatchNorm stats over N rows, 128 columns. Single block of 512 threads.
// ---------------------------------------------------------------------------
__global__ void bn_stats_kernel(const float* __restrict__ o, float* mu, float* rstd, int M) {
    __shared__ float ssum[512], ssq[512];
    int tid = threadIdx.x;
    int c = tid & 127, grp = tid >> 7;  // 4 row-groups
    float s = 0.0f, s2 = 0.0f;
    for (int r = grp; r < M; r += 4) {
        float val = o[(size_t)r * 128 + c];
        s += val; s2 += val * val;
    }
    ssum[tid] = s; ssq[tid] = s2;
    __syncthreads();
    if (grp == 0) {
        for (int g = 1; g < 4; ++g) { s += ssum[g * 128 + c]; s2 += ssq[g * 128 + c]; }
        float mn = s / (float)M;
        float var = s2 / (float)M - mn * mn;
        mu[c] = mn;
        rstd[c] = rsqrtf(var + 1e-5f);
    }
}

__global__ void bn_silu_add_kernel(const float* __restrict__ o, const float* __restrict__ mu,
                                   const float* __restrict__ rstd, const float* __restrict__ g,
                                   const float* __restrict__ b, const float* __restrict__ skip,
                                   float* __restrict__ nf, int n) {
    int i = blockIdx.x * 256 + threadIdx.x;
    if (i >= n) return;
    int c = i & 127;
    float val = (o[i] - mu[c]) * rstd[c] * g[c] + b[c];
    nf[i] = val * sigmoid_f(val) + skip[i];
}

__global__ void pool_sum_kernel(const float* __restrict__ nf, const int* __restrict__ batch,
                                float* pooled, float* cnt, int n) {
    int i = blockIdx.x * 256 + threadIdx.x;
    if (i >= n) return;
    int r = i >> 7, c = i & 127;
    int b = batch[r];
    atomicAdd(&pooled[b * 128 + c], nf[i]);
    if (c == 0) atomicAdd(&cnt[b], 1.0f);
}

__global__ void pool_feats_kernel(const float* __restrict__ pooled, const float* __restrict__ cnt,
                                  const float* __restrict__ pge, const float* __restrict__ cse,
                                  _Float16* __restrict__ feats) {
    int i = blockIdx.x * 256 + threadIdx.x;
    if (i >= 64 * 192) return;
    int b = i / 192, j = i - b * 192;
    float val;
    if (j < 128)      val = pooled[b * 128 + j] / fmaxf(cnt[b], 1.0f);
    else if (j < 160) val = pge[b * 32 + (j - 128)];
    else if (j < 167) val = cse[b * 7 + (j - 160)];
    else              val = 0.0f;
    feats[i] = (_Float16)val;
}

__global__ void silu_kernel(float* p, int n) {
    int i = blockIdx.x * 256 + threadIdx.x;
    if (i >= n) return;
    float v = p[i];
    p[i] = v * sigmoid_f(v);
}

__global__ void out_dot_kernel(const float* __restrict__ h, const float* __restrict__ w,
                               const float* __restrict__ ob, float* __restrict__ out) {
    int b = blockIdx.x;
    int tid = threadIdx.x;  // 128
    float s = 0.0f;
    for (int j = tid; j < 512; j += 128) s += h[b * 512 + j] * w[j];
    for (int o = 16; o > 0; o >>= 1) s += __shfl_xor(s, o);
    __shared__ float red[4];
    if ((tid & 31) == 0) red[tid >> 5] = s;
    __syncthreads();
    if (tid == 0) out[b] = red[0] + red[1] + red[2] + red[3] + ob[0];
}

// ---------------------------------------------------------------------------
// Host launcher
// ---------------------------------------------------------------------------
extern "C" void kernel_launch(void* const* d_in, const int* in_sizes, int n_in,
                              void* d_out, int out_size, void* d_ws, size_t ws_size,
                              hipStream_t stream) {
    (void)in_sizes; (void)n_in; (void)out_size; (void)ws_size;
    constexpr int N = 10000, E = 100000, B = 64, L = 2;

    const float* x         = (const float*)d_in[0];
    const float* edge_attr = (const float*)d_in[1];
    const float* pge       = (const float*)d_in[2];
    const float* cse       = (const float*)d_in[3];
    const int*   edge_idx  = (const int*)d_in[4];
    const int*   batch     = (const int*)d_in[5];
    const float* atom_W = (const float*)d_in[6];  const float* atom_b = (const float*)d_in[7];
    const float* edge_W = (const float*)d_in[8];  const float* edge_b = (const float*)d_in[9];
    const float* Wq = (const float*)d_in[10]; const float* bq = (const float*)d_in[11];
    const float* Wk = (const float*)d_in[12]; const float* bk = (const float*)d_in[13];
    const float* Wv = (const float*)d_in[14]; const float* bv = (const float*)d_in[15];
    const float* We = (const float*)d_in[16];
    const float* Wmu = (const float*)d_in[17]; const float* bmu = (const float*)d_in[18];
    const float* Wm  = (const float*)d_in[19]; const float* bm  = (const float*)d_in[20];
    const float* ln_m_g = (const float*)d_in[21]; const float* ln_m_b = (const float*)d_in[22];
    const float* ln_a_g = (const float*)d_in[23]; const float* ln_a_b = (const float*)d_in[24];
    const float* Wc = (const float*)d_in[25]; const float* bc = (const float*)d_in[26];
    const float* bn_g = (const float*)d_in[27]; const float* bn_b = (const float*)d_in[28];
    const float* Wskip = (const float*)d_in[29]; const float* bskip = (const float*)d_in[30];
    const float* fc_W = (const float*)d_in[31]; const float* fc_b = (const float*)d_in[32];
    const float* out_W = (const float*)d_in[33]; const float* out_b = (const float*)d_in[34];
    float* out = (float*)d_out;

    // ---- workspace carving ----
    char* ws = (char*)d_ws;
    size_t off = 0;
    auto carve = [&](size_t bytes) -> char* {
        char* p = ws + off;
        off += (bytes + 255) & ~(size_t)255;
        return p;
    };
    _Float16* xh    = (_Float16*)carve((size_t)N * 96 * 2);
    _Float16* eath  = (_Float16*)carve((size_t)E * 64 * 2);
    float*    nf    = (float*)carve((size_t)N * 128 * 4);
    _Float16* nfh   = (_Float16*)carve((size_t)N * 128 * 2);
    _Float16* efh   = (_Float16*)carve((size_t)E * 128 * 2);
    float*    qb    = (float*)carve((size_t)N * 256 * 4);
    float*    kb    = (float*)carve((size_t)N * 256 * 4);
    float*    vb    = (float*)carve((size_t)N * 256 * 4);
    _Float16* eah   = (_Float16*)carve((size_t)E * 256 * 2);
    float*    agg   = (float*)carve((size_t)N * 256 * 4);
    _Float16* aggh  = (_Float16*)carve((size_t)N * 256 * 2);
    float*    ob    = (float*)carve((size_t)N * 128 * 4);
    float*    skipb = (float*)carve((size_t)N * 128 * 4);
    float*    bn_mu = (float*)carve(128 * 4);
    float*    bn_rs = (float*)carve(128 * 4);
    float*    pooled= (float*)carve((size_t)B * 128 * 4);
    float*    cnt   = (float*)carve((size_t)B * 4);
    _Float16* featsh= (_Float16*)carve((size_t)B * 192 * 2);
    float*    hbuf  = (float*)carve((size_t)B * 512 * 4);
    _Float16* atomWT = (_Float16*)carve(128 * 96 * 2);
    _Float16* edgeWT = (_Float16*)carve(128 * 64 * 2);
    _Float16* WqT   = (_Float16*)carve((size_t)L * 256 * 128 * 2);
    _Float16* WkT   = (_Float16*)carve((size_t)L * 256 * 128 * 2);
    _Float16* WvT   = (_Float16*)carve((size_t)L * 256 * 128 * 2);
    _Float16* WeT   = (_Float16*)carve((size_t)L * 256 * 128 * 2);
    _Float16* WmuT  = (_Float16*)carve((size_t)L * 384 * 384 * 2);
    _Float16* WmT   = (_Float16*)carve((size_t)L * 128 * 384 * 2);
    _Float16* WcT   = (_Float16*)carve((size_t)L * 128 * 256 * 2);
    _Float16* WskipT= (_Float16*)carve((size_t)L * 128 * 128 * 2);
    _Float16* fcWT  = (_Float16*)carve(512 * 192 * 2);

    auto cdiv = [](int a, int b) { return (a + b - 1) / b; };

    // ---- weight conversion (f32 -> f16 transposed, K padded) ----
    convert_T_pad_kernel<<<cdiv(128 * 96, 256), 256, 0, stream>>>(atom_W, atomWT, 92, 128, 96);
    convert_T_pad_kernel<<<cdiv(128 * 64, 256), 256, 0, stream>>>(edge_W, edgeWT, 50, 128, 64);
    for (int l = 0; l < L; ++l) {
        convert_T_pad_kernel<<<cdiv(256 * 128, 256), 256, 0, stream>>>(Wq + (size_t)l * 128 * 256, WqT + (size_t)l * 256 * 128, 128, 256, 128);
        convert_T_pad_kernel<<<cdiv(256 * 128, 256), 256, 0, stream>>>(Wk + (size_t)l * 128 * 256, WkT + (size_t)l * 256 * 128, 128, 256, 128);
        convert_T_pad_kernel<<<cdiv(256 * 128, 256), 256, 0, stream>>>(Wv + (size_t)l * 128 * 256, WvT + (size_t)l * 256 * 128, 128, 256, 128);
        convert_T_pad_kernel<<<cdiv(256 * 128, 256), 256, 0, stream>>>(We + (size_t)l * 128 * 256, WeT + (size_t)l * 256 * 128, 128, 256, 128);
        convert_T_pad_kernel<<<cdiv(384 * 384, 256), 256, 0, stream>>>(Wmu + (size_t)l * 384 * 384, WmuT + (size_t)l * 384 * 384, 384, 384, 384);
        convert_T_pad_kernel<<<cdiv(128 * 384, 256), 256, 0, stream>>>(Wm + (size_t)l * 384 * 128, WmT + (size_t)l * 128 * 384, 384, 128, 384);
        convert_T_pad_kernel<<<cdiv(128 * 256, 256), 256, 0, stream>>>(Wc + (size_t)l * 256 * 128, WcT + (size_t)l * 128 * 256, 256, 128, 256);
        convert_T_pad_kernel<<<cdiv(128 * 128, 256), 256, 0, stream>>>(Wskip + (size_t)l * 128 * 128, WskipT + (size_t)l * 128 * 128, 128, 128, 128);
    }
    convert_T_pad_kernel<<<cdiv(512 * 192, 256), 256, 0, stream>>>(fc_W, fcWT, 167, 512, 192);

    // ---- input embedding ----
    convert_pad_rows_kernel<<<cdiv(N * 96, 256), 256, 0, stream>>>(x, xh, N, 92, 96);
    convert_pad_rows_kernel<<<cdiv(E * 64, 256), 256, 0, stream>>>(edge_attr, eath, E, 50, 64);
    gemm_wmma_kernel<float><<<dim3(N / 16, 2), 128, 0, stream>>>(xh, atomWT, atom_b, nf, N, 128, 96);
    gemm_wmma_kernel<_Float16><<<dim3(E / 16, 2), 128, 0, stream>>>(eath, edgeWT, edge_b, efh, E, 128, 64);

    // ---- layers ----
    for (int l = 0; l < L; ++l) {
        f32_to_f16_kernel<<<cdiv(N * 128, 256), 256, 0, stream>>>(nf, nfh, N * 128);
        gemm_wmma_kernel<float><<<dim3(N / 16, 4), 128, 0, stream>>>(nfh, WqT + (size_t)l * 256 * 128, bq + l * 256, qb, N, 256, 128);
        gemm_wmma_kernel<float><<<dim3(N / 16, 4), 128, 0, stream>>>(nfh, WkT + (size_t)l * 256 * 128, bk + l * 256, kb, N, 256, 128);
        gemm_wmma_kernel<float><<<dim3(N / 16, 4), 128, 0, stream>>>(nfh, WvT + (size_t)l * 256 * 128, bv + l * 256, vb, N, 256, 128);
        gemm_wmma_kernel<_Float16><<<dim3(E / 16, 4), 128, 0, stream>>>(efh, WeT + (size_t)l * 256 * 128, (const float*)nullptr, eah, E, 256, 128);

        zero_kernel<<<cdiv(N * 256, 256), 256, 0, stream>>>(agg, N * 256);
        edge_msg_kernel<<<(2 * E) / 32, 256, 114688, stream>>>(
            qb, kb, vb, eah, edge_idx,
            WmuT + (size_t)l * 384 * 384, bmu + l * 384,
            WmT + (size_t)l * 128 * 384, bm + l * 128,
            ln_a_g + l * 384, ln_a_b + l * 384,
            ln_m_g + l * 128, ln_m_b + l * 128,
            agg, E);

        f32_to_f16_kernel<<<cdiv(N * 256, 256), 256, 0, stream>>>(agg, aggh, N * 256);
        gemm_wmma_kernel<float><<<dim3(N / 16, 2), 128, 0, stream>>>(aggh, WcT + (size_t)l * 128 * 256, bc + l * 128, ob, N, 128, 256);
        bn_stats_kernel<<<1, 512, 0, stream>>>(ob, bn_mu, bn_rs, N);
        gemm_wmma_kernel<float><<<dim3(N / 16, 2), 128, 0, stream>>>(nfh, WskipT + (size_t)l * 128 * 128, bskip + l * 128, skipb, N, 128, 128);
        bn_silu_add_kernel<<<cdiv(N * 128, 256), 256, 0, stream>>>(ob, bn_mu, bn_rs, bn_g + l * 128, bn_b + l * 128, skipb, nf, N * 128);
    }

    // ---- readout ----
    zero_kernel<<<cdiv(B * 128, 256), 256, 0, stream>>>(pooled, B * 128);
    zero_kernel<<<1, 256, 0, stream>>>(cnt, B);
    pool_sum_kernel<<<cdiv(N * 128, 256), 256, 0, stream>>>(nf, batch, pooled, cnt, N * 128);
    pool_feats_kernel<<<cdiv(B * 192, 256), 256, 0, stream>>>(pooled, cnt, pge, cse, featsh);
    gemm_wmma_kernel<float><<<dim3(B / 16, 8), 128, 0, stream>>>(featsh, fcWT, fc_b, hbuf, B, 512, 192);
    silu_kernel<<<cdiv(B * 512, 256), 256, 0, stream>>>(hbuf, B * 512);
    out_dot_kernel<<<B, 128, 0, stream>>>(hbuf, out_W, out_b, out);
}
